// Model_38903813767339
// MI455X (gfx1250) — matrix operand order repaired and checked by reading, not compiled, not run on hardware
//
#include <hip/hip_runtime.h>
#include <hip/hip_bf16.h>
#include <stdint.h>

#define BB 4
#define NN 64
#define TT 64
#define DD 128
#define EE 4
#define HH 32
#define BT (BB * TT)

typedef __attribute__((ext_vector_type(16))) _Float16 v16h;
typedef __attribute__((ext_vector_type(8)))  float    v8f;
typedef __attribute__((ext_vector_type(4)))  unsigned u32x4;
typedef __attribute__((ext_vector_type(8)))  unsigned u32x8;

// ---- LDS layout (bytes) ----
#define OFF_ZH    0u        // f16 [64][128]   16384
#define OFF_Q     16384u    // f16 [64][128]   16384   (aliased later: f32 out buffer spans Q+K)
#define OFF_K     32768u    // f16 [64][128]   16384
#define OFF_V     49152u    // f16 [64][128]   16384
#define OFF_S     65536u    // f32 [64][64]    16384
#define OFF_ALPHA 81920u    // f16 [64][64]     8192
#define OFF_SP    90112u    // f16 [64][128]   16384
#define OFF_W     106496u   // f16 [128][128]  32768  (also holds We1[:, :256] as [32][256])
#define OFF_QH    139264u   // f32 [64][32]     8192
#define OFF_KH    147456u   // f32 [64][32]     8192
#define OFF_EW    155648u   // f32 [32][4]       512
#define OFF_BE1   156160u   // f32 [32]
#define OFF_WE2   156288u   // f32 [32]
#define OFF_LNW   156416u   // f32 [128]
#define OFF_LNB   156928u   // f32 [128]
#define OFF_WFUSE 157440u   // f32 [8]
#define OFF_MASK  157472u   // u8  [64]
#define OFF_OUT   16384u    // f32 [64][128] 32768 (alias over Q,K after they die)
#define SMEM_BYTES 157568u

// ws layout: Zh f16 [BT][64][128] then weight pack f16 (Wq,Wk,Wv,Wtheta each 16384h, We1qk 8192h)
#define WS_ZH_HALVES   ((size_t)BT * NN * DD)          // 2,097,152 halves
#define WS_W_HALVES    (4 * DD * DD + HH * 2 * DD)     // 73,728 halves
#define WS_NEED_BYTES  ((WS_ZH_HALVES + WS_W_HALVES) * 2)

// ---------------- WMMA fragment helpers (A/B: contiguous k = 16*(lane/16)+s; C/D per ISA 7.12.2) ----------------

// A matrix 16x32 f16. Lanes 0-15: M=lane, K=0..15; lanes 16-31: M=lane-16, K=16..31
__device__ __forceinline__ v16h ldsA(const _Float16* src, int ld) {
  const int lane = threadIdx.x & 31;
  const int m = lane & 15;
  const int hi = lane >> 4;
  const _Float16* p = src + m * ld;
  v16h a;
#pragma unroll
  for (int v = 0; v < 8; ++v) {
    int kb = 2 * v + (hi ? 16 : 0);
    a[2 * v]     = p[kb];
    a[2 * v + 1] = p[kb + 1];
  }
  return a;
}

// B matrix 32x16 f16 where B[k][n] = srcW[n*ld + k] (B = srcW^T, srcW row-major [n][k])
__device__ __forceinline__ v16h ldsB_nt(const _Float16* srcW, int ld) {
  const int lane = threadIdx.x & 31;
  const int n = lane & 15;
  const int hi = lane >> 4;
  const _Float16* p = srcW + n * ld + (hi ? 16 : 0);
  v16h b;
#pragma unroll
  for (int v = 0; v < 8; ++v) {
    b[2 * v]     = p[2 * v];
    b[2 * v + 1] = p[2 * v + 1];
  }
  return b;
}

// B matrix 32x16 f16 where B[k][n] = src[k*ld + n] (row-major source)
__device__ __forceinline__ v16h ldsB_n(const _Float16* src, int ld) {
  const int lane = threadIdx.x & 31;
  const int n = lane & 15;
  const int hi = lane >> 4;
  const _Float16* p = src + (hi ? 16 : 0) * ld + n;
  v16h b;
#pragma unroll
  for (int v = 0; v < 8; ++v) {
    b[2 * v]     = p[(2 * v) * ld];
    b[2 * v + 1] = p[(2 * v + 1) * ld];
  }
  return b;
}

// C/D f32 16x16: VGPR r, lanes 0-15 -> (M=r, N=lane); lanes 16-31 -> (M=r+8, N=lane-16)
__device__ __forceinline__ void stC_f16(_Float16* dst, int ld, v8f c) {
  const int lane = threadIdx.x & 31;
  const int n = lane & 15;
  const int hi = lane >> 4;
#pragma unroll
  for (int r = 0; r < 8; ++r) dst[(r + (hi ? 8 : 0)) * ld + n] = (_Float16)c[r];
}

__device__ __forceinline__ void stC_f32(float* dst, int ld, v8f c, float scale) {
  const int lane = threadIdx.x & 31;
  const int n = lane & 15;
  const int hi = lane >> 4;
#pragma unroll
  for (int r = 0; r < 8; ++r) dst[(r + (hi ? 8 : 0)) * ld + n] = c[r] * scale;
}

#define WMMA_F16(a, b, c) \
  __builtin_amdgcn_wmma_f32_16x16x32_f16(false, (a), false, (b), (short)0, (c), false, false)

// ---------------- CDNA5 async / TDM data movement ----------------

// Async global->LDS copy, 16B per lane per op (GLOBAL_LOAD_ASYNC_TO_LDS_B128, ASYNCcnt)
__device__ __forceinline__ void async_load_lds(const _Float16* __restrict__ g,
                                               _Float16* s, int halves) {
  const unsigned lds0 = (unsigned)(uintptr_t)s;  // low 32 bits of flat shared addr == LDS addr
  const char* gp = (const char*)g;
  const int bytes = halves * 2;
  for (int off = threadIdx.x * 16; off < bytes; off += 256 * 16) {
    asm volatile("global_load_async_to_lds_b128 %0, %1, off"
                 :: "v"(lds0 + (unsigned)off),
                    "v"((unsigned long long)(uintptr_t)(gp + off))
                 : "memory");
  }
  asm volatile("s_wait_asynccnt 0" ::: "memory");
}

// Tensor Data Mover: 1-D tile of `elems` 2-byte elements, global -> LDS (TENSORcnt)
__device__ __forceinline__ void tdm_load_1d(unsigned lds_addr, const void* gptr, unsigned elems) {
  const unsigned long long ga = (unsigned long long)(uintptr_t)gptr;
  u32x4 g0;
  g0[0] = 1u;                                            // count=1, user mode
  g0[1] = lds_addr;                                      // lds_addr [63:32]
  g0[2] = (unsigned)(ga & 0xFFFFFFFFu);                  // global_addr lo
  g0[3] = (unsigned)((ga >> 32) & 0x1FFFFFFu) | (2u << 30);  // global_addr[56:32] | type=2
  u32x8 g1;
  g1[0] = 0x00010000u;                                   // data_size=1 (2 bytes)
  g1[1] = (elems & 0xFFFFu) << 16;                       // tensor_dim0[15:0]
  g1[2] = ((elems >> 16) & 0xFFFFu) | (1u << 16);        // tensor_dim0 hi | tensor_dim1=1
  g1[3] = (elems & 0xFFFFu) << 16;                       // tile_dim0 = elems (<=65535)
  g1[4] = 1u;                                            // tile_dim1=1, tile_dim2=0
  g1[5] = elems;                                         // tensor_dim0_stride lo
  g1[6] = 0u;
  g1[7] = 0u;
  asm volatile("tensor_load_to_lds %0, %1" :: "s"(g0), "s"(g1) : "memory");
}

__global__ __launch_bounds__(256) void prep_weights_kernel(
    const float* __restrict__ Wq, const float* __restrict__ Wk,
    const float* __restrict__ Wv, const float* __restrict__ Wtheta,
    const float* __restrict__ We1, _Float16* __restrict__ wf16) {
  const int idx = blockIdx.x * 256 + threadIdx.x;
  if (idx < 4 * DD * DD) {
    const int which = idx >> 14, k = idx & 16383;
    const float* src = which == 0 ? Wq : which == 1 ? Wk : which == 2 ? Wv : Wtheta;
    wf16[idx] = (_Float16)src[k];
  } else if (idx < 4 * DD * DD + HH * 2 * DD) {
    const int i = idx - 4 * DD * DD;
    const int h = i >> 8, k = i & 255;
    wf16[idx] = (_Float16)We1[h * (2 * DD + EE) + k];  // [32][256] f16
  }
}

__global__ __launch_bounds__(256) void prep_z_kernel(const float* __restrict__ x,
                                                     _Float16* __restrict__ zh) {
  // zh[((b*T+t)*N + n)*D + d] = x[((b*N+n)*T + t)*D + d]
  const int idx = blockIdx.x * 256 + threadIdx.x;  // over B*N*T*(D/4)
  if (idx >= BB * NN * TT * (DD / 4)) return;
  const int d4 = idx & 31;
  int rest = idx >> 5;
  const int t = rest & 63; rest >>= 6;
  const int n = rest & 63;
  const int b = rest >> 6;
  const float4 v = *(const float4*)(x + (((size_t)(b * NN + n) * TT + t) * DD) + d4 * 4);
  _Float16* dst = zh + (((size_t)(b * TT + t) * NN + n) * DD) + d4 * 4;
  dst[0] = (_Float16)v.x; dst[1] = (_Float16)v.y;
  dst[2] = (_Float16)v.z; dst[3] = (_Float16)v.w;
}

// C[64x128] = A[64x128](f16) * W^T  (W row-major [128][128] f16 in LDS), f16 result
__device__ __forceinline__ void gemm_proj(const _Float16* A, const _Float16* Wlds, _Float16* Cout) {
  const int wave = threadIdx.x >> 5;
  for (int tile = wave; tile < 32; tile += 8) {
    int mt = tile >> 3, nt = tile & 7;
    v8f c = {};
#pragma unroll
    for (int kk = 0; kk < DD; kk += 32) {
      v16h a = ldsA(A + mt * 16 * DD + kk, DD);
      v16h b = ldsB_nt(Wlds + nt * 16 * DD + kk, DD);
      c = WMMA_F16(a, b, c);
    }
    stC_f16(Cout + mt * 16 * DD + nt * 16, DD, c);
  }
}

__global__ __launch_bounds__(256) void stgat_fused_kernel(
    const float* __restrict__ x, const float* __restrict__ edge_features,
    const float* __restrict__ A_prior, const uint8_t* __restrict__ entity_mask,
    const float* __restrict__ W_fuse, const float* __restrict__ Wq,
    const float* __restrict__ Wk, const float* __restrict__ Wv,
    const float* __restrict__ We1, const float* __restrict__ be1,
    const float* __restrict__ We2, const float* __restrict__ be2,
    const float* __restrict__ Wtheta, const float* __restrict__ ln_w,
    const float* __restrict__ ln_b, const float* __restrict__ phys_w,
    const float* __restrict__ prior_w,
    const _Float16* __restrict__ zh_ws,   // nullptr -> fallback conversion path
    const _Float16* __restrict__ wf16_ws, // nullptr -> fallback conversion path
    float* __restrict__ out)
{
  __shared__ __align__(16) unsigned char smem[SMEM_BYTES];
  _Float16* sZ     = (_Float16*)(smem + OFF_ZH);
  _Float16* sQ     = (_Float16*)(smem + OFF_Q);
  _Float16* sK     = (_Float16*)(smem + OFF_K);
  _Float16* sV     = (_Float16*)(smem + OFF_V);
  float*    sS     = (float*)(smem + OFF_S);
  _Float16* sAlpha = (_Float16*)(smem + OFF_ALPHA);
  _Float16* sSp    = (_Float16*)(smem + OFF_SP);
  _Float16* sW     = (_Float16*)(smem + OFF_W);     // weight staging (reused)
  float*    sQh    = (float*)(smem + OFF_QH);
  float*    sKh    = (float*)(smem + OFF_KH);
  float*    sEW    = (float*)(smem + OFF_EW);
  float*    sBE1   = (float*)(smem + OFF_BE1);
  float*    sWE2   = (float*)(smem + OFF_WE2);
  float*    sLNW   = (float*)(smem + OFF_LNW);
  float*    sLNB   = (float*)(smem + OFF_LNB);
  float*    sWF    = (float*)(smem + OFF_WFUSE);
  uint8_t*  sMask  = (uint8_t*)(smem + OFF_MASK);

  const int tid  = threadIdx.x;
  const int wave = tid >> 5;
  const int lane = tid & 31;
  const int bt = blockIdx.x;
  const int b = bt / TT, t = bt % TT;

  // ---- Stage A: Z tile into LDS (TDM DMA when staged) + small params ----
  if (zh_ws) {
    if (wave == 0) {
      tdm_load_1d((unsigned)(uintptr_t)sZ, zh_ws + (size_t)bt * NN * DD, NN * DD);
    }
    __builtin_amdgcn_s_wait_tensorcnt(0);
  } else {
    for (int idx = tid; idx < NN * DD; idx += 256) {
      int n = idx >> 7, d = idx & 127;
      sZ[idx] = (_Float16)x[(((size_t)(b * NN + n) * TT + t) * DD) + d];
    }
  }
  if (tid < HH * EE) { int h = tid >> 2, e = tid & 3; sEW[tid] = We1[h * (2 * DD + EE) + 2 * DD + e]; }
  if (tid < HH) { sBE1[tid] = be1[tid]; sWE2[tid] = We2[tid]; }
  if (tid < DD) { sLNW[tid] = ln_w[tid]; sLNB[tid] = ln_b[tid]; }
  if (tid < 5)  sWF[tid] = W_fuse[tid];
  if (tid < NN) sMask[tid] = entity_mask[b * NN + tid];
  __syncthreads();

  // ---- Stage B: Q, K, V projections (WMMA f16, f32 accumulate) ----
#define STAGE_W(SRC_F32, WOFF)                                              \
  if (wf16_ws) async_load_lds(wf16_ws + (WOFF), sW, DD * DD);               \
  else { for (int i = tid; i < DD * DD; i += 256) sW[i] = (_Float16)SRC_F32[i]; }

  STAGE_W(Wq, 0);          __syncthreads();
  gemm_proj(sZ, sW, sQ);   __syncthreads();
  STAGE_W(Wk, DD * DD);    __syncthreads();
  gemm_proj(sZ, sW, sK);   __syncthreads();
  STAGE_W(Wv, 2 * DD * DD); __syncthreads();
  gemm_proj(sZ, sW, sV);   __syncthreads();

  // ---- Stage C: content logits S = Q K^T / sqrt(D) ----
  {
    const float scale = 0.08838834764831845f;  // 1/sqrt(128)
    for (int tile = wave; tile < 16; tile += 8) {
      int mt = tile >> 2, nt = tile & 3;
      v8f c = {};
#pragma unroll
      for (int kk = 0; kk < DD; kk += 32) {
        v16h a = ldsA(sQ + mt * 16 * DD + kk, DD);
        v16h bb = ldsB_nt(sK + nt * 16 * DD + kk, DD);  // B[k][j] = K[j][k]
        c = WMMA_F16(a, bb, c);
      }
      stC_f32(sS + mt * 16 * NN + nt * 16, NN, c, scale);
    }
  }
  __syncthreads();

  // ---- Stage D: qh = Q @ We1[:, :D]^T, kh = K @ We1[:, D:2D]^T ----
  if (wf16_ws) {
    async_load_lds(wf16_ws + 4 * DD * DD, sW, HH * 2 * DD);
  } else {
    for (int i = tid; i < HH * 2 * DD; i += 256) {
      int h = i >> 8, k = i & 255;
      sW[h * 256 + k] = (_Float16)We1[h * (2 * DD + EE) + k];  // [32][256] f16
    }
  }
  __syncthreads();
  for (int tile = wave; tile < 16; tile += 8) {
    const bool isK = tile >= 8;
    int tt = tile & 7;
    int mt = tt >> 1, nt = tt & 1;
    const _Float16* A = isK ? sK : sQ;
    const int off = isK ? DD : 0;
    v8f c = {};
#pragma unroll
    for (int kk = 0; kk < DD; kk += 32) {
      v16h a = ldsA(A + mt * 16 * DD + kk, DD);
      v16h bb = ldsB_nt(sW + nt * 16 * 256 + off + kk, 256);  // B[k][h] = We1[h][off+k]
      c = WMMA_F16(a, bb, c);
    }
    stC_f32((isK ? sKh : sQh) + mt * 16 * HH + nt * 16, HH, c, 1.0f);
  }
  __syncthreads();

  // ---- Stage E: edge MLP + prior fuse + mask, accumulate into sS ----
  {
    const float pw = phys_w[0], rw = prior_w[0], be2v = be2[0];
    for (int p = tid; p < NN * NN; p += 256) {
      int i = p >> 6, j = p & 63;
      const float* ef = edge_features + ((size_t)bt * NN * NN + p) * EE;
      __builtin_prefetch(ef + 256, 0, 0);
      float e0 = ef[0], e1 = ef[1], e2 = ef[2], e3 = ef[3];
      float acc = 0.f;
#pragma unroll
      for (int h = 0; h < HH; ++h) {
        float v = sQh[i * HH + h] + sKh[j * HH + h] + sBE1[h]
                + e0 * sEW[h * 4 + 0] + e1 * sEW[h * 4 + 1]
                + e2 * sEW[h * 4 + 2] + e3 * sEW[h * 4 + 3];
        acc += fmaxf(v, 0.f) * sWE2[h];
      }
      float logit = sS[p] + pw * (acc + be2v);
      const float* ap = A_prior + ((size_t)bt * NN * NN + p) * 5;
      float apv = ap[0] * sWF[0] + ap[1] * sWF[1] + ap[2] * sWF[2]
                + ap[3] * sWF[3] + ap[4] * sWF[4];
      if (!(apv == apv)) apv = 0.f;     // nan_to_num
      apv = fmaxf(apv, 0.f);
      logit += rw * logf(apv + 1e-6f);
      if (sMask[i] | sMask[j]) logit = -1e9f;
      sS[p] = logit;
    }
  }
  __syncthreads();

  // ---- Stage F: row softmax (wave-level shuffle reductions), alpha -> f16 ----
  for (int i = wave; i < NN; i += 8) {
    float v0 = sS[i * NN + lane], v1 = sS[i * NN + lane + 32];
    float m = fmaxf(v0, v1);
#pragma unroll
    for (int o = 16; o > 0; o >>= 1) m = fmaxf(m, __shfl_xor(m, o, 32));
    float e0v = __expf(v0 - m), e1v = __expf(v1 - m);
    float s = e0v + e1v;
#pragma unroll
    for (int o = 16; o > 0; o >>= 1) s += __shfl_xor(s, o, 32);
    float inv = 1.0f / s;
    sAlpha[i * NN + lane]      = (_Float16)(e0v * inv);
    sAlpha[i * NN + lane + 32] = (_Float16)(e1v * inv);
  }
  __syncthreads();
  STAGE_W(Wtheta, 3 * DD * DD);
  __syncthreads();

  // ---- Stage G: spatial = alpha @ V ----
  for (int tile = wave; tile < 32; tile += 8) {
    int mt = tile >> 3, nt = tile & 7;
    v8f c = {};
#pragma unroll
    for (int kk = 0; kk < NN; kk += 32) {
      v16h a = ldsA(sAlpha + mt * 16 * NN + kk, NN);
      v16h bb = ldsB_n(sV + kk * DD + nt * 16, DD);  // B[k][d] = V[k][d]
      c = WMMA_F16(a, bb, c);
    }
    stC_f16(sSp + mt * 16 * DD + nt * 16, DD, c);
  }
  __syncthreads();

  // ---- Stage H: out = spatial @ Wtheta^T + Z (residual, f32, aliased over dead Q/K) ----
  {
    float* sOut = (float*)(smem + OFF_OUT);
    for (int tile = wave; tile < 32; tile += 8) {
      int mt = tile >> 3, nt = tile & 7;
      v8f c = {};
#pragma unroll
      for (int kk = 0; kk < DD; kk += 32) {
        v16h a = ldsA(sSp + mt * 16 * DD + kk, DD);
        v16h bb = ldsB_nt(sW + nt * 16 * DD + kk, DD);
        c = WMMA_F16(a, bb, c);
      }
      const int n2 = lane & 15, hi = lane >> 4;
#pragma unroll
      for (int r = 0; r < 8; ++r) {
        int row = mt * 16 + r + (hi ? 8 : 0);
        int col = nt * 16 + n2;
        sOut[row * DD + col] = c[r] + (float)sZ[row * DD + col];
      }
    }
  }
  __syncthreads();

  // ---- Stage I: layernorm + entity-mask zero + transposed store ----
  {
    const float* sOut = (const float*)(smem + OFF_OUT);
    for (int r = wave; r < NN; r += 8) {
      const int d0 = lane * 4;
      const float4 v = *(const float4*)(sOut + r * DD + d0);
      float s1 = v.x + v.y + v.z + v.w;
      float s2 = v.x * v.x + v.y * v.y + v.z * v.z + v.w * v.w;
#pragma unroll
      for (int o = 16; o > 0; o >>= 1) {
        s1 += __shfl_xor(s1, o, 32);
        s2 += __shfl_xor(s2, o, 32);
      }
      float mu = s1 * (1.0f / DD);
      float var = s2 * (1.0f / DD) - mu * mu;
      float rs = rsqrtf(var + 1e-5f);
      const bool msk = sMask[r] != 0;
      float4 o4;
      o4.x = msk ? 0.f : (v.x - mu) * rs * sLNW[d0 + 0] + sLNB[d0 + 0];
      o4.y = msk ? 0.f : (v.y - mu) * rs * sLNW[d0 + 1] + sLNB[d0 + 1];
      o4.z = msk ? 0.f : (v.z - mu) * rs * sLNW[d0 + 2] + sLNB[d0 + 2];
      o4.w = msk ? 0.f : (v.w - mu) * rs * sLNW[d0 + 3] + sLNB[d0 + 3];
      *(float4*)(out + (((size_t)(b * NN + r) * TT + t) * DD) + d0) = o4;
    }
  }
}

extern "C" void kernel_launch(void* const* d_in, const int* in_sizes, int n_in,
                              void* d_out, int out_size, void* d_ws, size_t ws_size,
                              hipStream_t stream) {
  (void)in_sizes; (void)n_in; (void)out_size;
  const float*   x      = (const float*)d_in[0];
  const float*   ef     = (const float*)d_in[1];
  const float*   ap     = (const float*)d_in[2];
  const uint8_t* em     = (const uint8_t*)d_in[3];
  const float*   wfuse  = (const float*)d_in[4];
  const float*   wq     = (const float*)d_in[5];
  const float*   wk     = (const float*)d_in[6];
  const float*   wv     = (const float*)d_in[7];
  const float*   we1    = (const float*)d_in[8];
  const float*   be1    = (const float*)d_in[9];
  const float*   we2    = (const float*)d_in[10];
  const float*   be2    = (const float*)d_in[11];
  const float*   wtheta = (const float*)d_in[12];
  const float*   lnw    = (const float*)d_in[13];
  const float*   lnb    = (const float*)d_in[14];
  const float*   pw     = (const float*)d_in[15];
  const float*   rw     = (const float*)d_in[16];

  const bool use_ws = (d_ws != nullptr) && (ws_size >= WS_NEED_BYTES);
  _Float16* zh   = use_ws ? (_Float16*)d_ws : nullptr;
  _Float16* wf16 = use_ws ? (_Float16*)((char*)d_ws + WS_ZH_HALVES * 2) : nullptr;

  if (use_ws) {
    prep_z_kernel<<<(BB * NN * TT * (DD / 4) + 255) / 256, 256, 0, stream>>>(x, zh);
    prep_weights_kernel<<<(WS_W_HALVES + 255) / 256, 256, 0, stream>>>(wq, wk, wv, wtheta, we1, wf16);
  }
  stgat_fused_kernel<<<BT, 256, 0, stream>>>(x, ef, ap, em, wfuse, wq, wk, wv,
                                             we1, be1, we2, be2, wtheta, lnw, lnb,
                                             pw, rw, zh, wf16, (float*)d_out);
}
